// HYV3Attention_90099823935490
// MI455X (gfx1250) — matrix operand order repaired and checked by reading, not compiled, not run on hardware
//
#include <hip/hip_runtime.h>

// ---------------- CDNA5 WMMA types ----------------
typedef __attribute__((ext_vector_type(16))) __bf16 v16bf;
typedef __attribute__((ext_vector_type(8)))  float  v8f;

union Frag16 { unsigned u[8]; v16bf v; };   // 16x32 (A) or 32x16 (B) bf16 fragment
union Pack2  { __bf16 h[2]; unsigned u; };
union Pack4  { __bf16 h[4]; uint2 u2; };

static __device__ __forceinline__ v8f wmma_bf16(const Frag16& a, const Frag16& b, v8f c) {
    return __builtin_amdgcn_wmma_f32_16x16x32_bf16(
        /*neg_a=*/false, a.v, /*neg_b=*/false, b.v,
        /*c_mod=*/(short)0, c, /*reuse_a=*/false, /*reuse_b=*/false);
}

// ---------------- CDNA5 async global->LDS DMA (ASYNCcnt path) ----------------
#if __has_builtin(__builtin_amdgcn_global_load_async_to_lds_b128)
#define HAVE_ASYNC_LDS 1
typedef int v4i_vec __attribute__((vector_size(16)));              // int __vector(4)
typedef __attribute__((address_space(1))) v4i_vec* gptr_b128;      // global (prints __device__)
typedef __attribute__((address_space(3))) v4i_vec* lptr_b128;      // LDS    (prints __shared__)
#else
#define HAVE_ASYNC_LDS 0
#endif

static __device__ __forceinline__ void async_ld_b128(const void* g, void* l) {
#if HAVE_ASYNC_LDS
    __builtin_amdgcn_global_load_async_to_lds_b128((gptr_b128)g, (lptr_b128)l, 0, 0);
#else
    *reinterpret_cast<uint4*>(l) = *reinterpret_cast<const uint4*>(g);
#endif
}

static __device__ __forceinline__ void wait_async0() {
#if HAVE_ASYNC_LDS
#if __has_builtin(__builtin_amdgcn_s_wait_asynccnt)
    __builtin_amdgcn_s_wait_asynccnt(0);
#else
    asm volatile("s_wait_asynccnt 0x0" ::: "memory");
#endif
#endif
}

// ---------------- problem constants ----------------
constexpr int TT      = 2048;
constexpr int HIDDEN  = 4096;
constexpr int NH      = 32;
constexpr int NKV     = 8;
constexpr int HD      = 128;
constexpr int QKV_OUT = (NH + 2 * NKV) * HD;   // 6144
constexpr float RMS_EPS = 1e-5f;

// =====================================================================
// fp32 -> bf16 conversion (grid-stride, 4-wide)
// =====================================================================
__global__ void cvt_f32_bf16(const float* __restrict__ src, __bf16* __restrict__ dst, int n4) {
    int i = blockIdx.x * blockDim.x + threadIdx.x;
    int stride = gridDim.x * blockDim.x;
    for (; i < n4; i += stride) {
        float4 v = reinterpret_cast<const float4*>(src)[i];
        Pack4 p;
        p.h[0] = (__bf16)v.x; p.h[1] = (__bf16)v.y;
        p.h[2] = (__bf16)v.z; p.h[3] = (__bf16)v.w;
        reinterpret_cast<uint2*>(dst)[i] = p.u2;
    }
}

// =====================================================================
// C(MxN, f32) = A(MxK, bf16 row-major) * B(NxK, bf16 row-major)^T
// 128x128 block tile, BK=32, 8 waves; each wave: 2x4 grid of 16x16 WMMAs.
// Staging via async global->LDS DMA (no VGPR bounce).
// =====================================================================
constexpr int BM = 128, BN = 128, BK = 32;

__global__ __launch_bounds__(256) void gemm_bf16_nt(
    const __bf16* __restrict__ A, const __bf16* __restrict__ B,
    float* __restrict__ C, int M, int N, int K)
{
    __shared__ __bf16 sA[BM][BK];
    __shared__ __bf16 sB[BN][BK];

    const int tid  = threadIdx.x;
    const int lane = tid & 31;
    const int wave = tid >> 5;
    const int r    = lane & 15;     // row/col within 16x16 subtile
    const int kg   = lane >> 4;     // k-group select
    const int bm   = blockIdx.y * BM;
    const int bn   = blockIdx.x * BN;
    const int wm   = (wave & 3) * 32;   // wave M offset in tile
    const int wn   = (wave >> 2) * 64;  // wave N offset in tile

    v8f acc[2][4];
#pragma unroll
    for (int i = 0; i < 2; i++)
#pragma unroll
        for (int j = 0; j < 4; j++)
            acc[i][j] = v8f{0.f,0.f,0.f,0.f,0.f,0.f,0.f,0.f};

    // staging coords: each of 256 threads moves 16 bf16 per tile
    const int ar = tid >> 1;
    const int ac = (tid & 1) * 16;

    for (int kk = 0; kk < K; kk += BK) {
        const __bf16* gA = A + (size_t)(bm + ar) * K + kk + ac;
        const __bf16* gB = B + (size_t)(bn + ar) * K + kk + ac;
        async_ld_b128(gA,     &sA[ar][ac]);
        async_ld_b128(gA + 8, &sA[ar][ac + 8]);
        async_ld_b128(gB,     &sB[ar][ac]);
        async_ld_b128(gB + 8, &sB[ar][ac + 8]);
        __builtin_prefetch(gA + BK, 0, 1);   // next K tile -> global_prefetch
        __builtin_prefetch(gB + BK, 0, 1);
        wait_async0();
        __syncthreads();

        // ---- fragment gather per CDNA5 16-bit layouts (pair = one u32) ----
        const unsigned* lA = reinterpret_cast<const unsigned*>(sA);  // row stride 16 u32
        const unsigned* lB = reinterpret_cast<const unsigned*>(sB);

        Frag16 af[2], bf[4];
#pragma unroll
        for (int i = 0; i < 2; i++) {
            const int row = wm + i * 16 + r;
#pragma unroll
            for (int j = 0; j < 4; j++) {
                af[i].u[j]     = lA[row * 16 + kg * 4 + j];      // K = kg*8+2j
                af[i].u[4 + j] = lA[row * 16 + 8 + kg * 4 + j];  // K = 16+kg*8+2j
            }
        }
#pragma unroll
        for (int j = 0; j < 4; j++) {
            const int rowB = wn + j * 16 + r;                    // N index = B row
#pragma unroll
            for (int i2 = 0; i2 < 8; i2++)
                bf[j].u[i2] = lB[rowB * 16 + (i2 >> 2) * 8 + kg * 4 + (i2 & 3)];  // K = (i2>>2)*16+kg*8+2*(i2&3)
        }
#pragma unroll
        for (int i = 0; i < 2; i++)
#pragma unroll
            for (int j = 0; j < 4; j++)
                acc[i][j] = wmma_bf16(af[i], bf[j], acc[i][j]);
        __syncthreads();
    }

    // ---- store C (f32 C-layout: VGPR rr -> M = rr + 8*kg, N = lane%16) ----
#pragma unroll
    for (int i = 0; i < 2; i++)
#pragma unroll
        for (int j = 0; j < 4; j++) {
            const int row0 = bm + wm + i * 16 + kg * 8;
            const int col  = bn + wn + j * 16 + r;
#pragma unroll
            for (int rr = 0; rr < 8; rr++)
                C[(size_t)(row0 + rr) * N + col] = acc[i][j][rr];
        }
}

// =====================================================================
// Fused per-head RMSNorm + NeoX RoPE + bf16 pack.
// One wave per (token, slot); slots: 0..31 q-heads, 32..39 k, 40..47 v.
// Outputs head-major: q[NH][T][HD], k[NKV][T][HD], v[NKV][T][HD].
// =====================================================================
__global__ __launch_bounds__(128) void norm_rope_bf16(
    const float* __restrict__ qkv, const long long* __restrict__ pos,
    const float* __restrict__ qw, const float* __restrict__ kw,
    __bf16* __restrict__ qb, __bf16* __restrict__ kb, __bf16* __restrict__ vb)
{
    const int wid  = blockIdx.x * 4 + (threadIdx.x >> 5);
    const int lane = threadIdx.x & 31;
    const int t    = wid / 48;
    const int slot = wid % 48;

    const float* x = qkv + (size_t)t * QKV_OUT + slot * HD;
    float4 xv = reinterpret_cast<const float4*>(x)[lane];   // d = lane*4 .. +3

    if (slot >= 40) {   // V: straight bf16 convert
        Pack4 p;
        p.h[0] = (__bf16)xv.x; p.h[1] = (__bf16)xv.y;
        p.h[2] = (__bf16)xv.z; p.h[3] = (__bf16)xv.w;
        __bf16* dst = vb + ((size_t)(slot - 40) * TT + t) * HD + lane * 4;
        *reinterpret_cast<uint2*>(dst) = p.u2;
        return;
    }

    // RMS norm over 128
    float ss = xv.x*xv.x + xv.y*xv.y + xv.z*xv.z + xv.w*xv.w;
#pragma unroll
    for (int m = 1; m < 32; m <<= 1) ss += __shfl_xor(ss, m, 32);
    const float rstd = rsqrtf(ss * (1.0f / HD) + RMS_EPS);

    const float* w = (slot < 32) ? qw : kw;
    float xn[4];
    float xin[4] = {xv.x, xv.y, xv.z, xv.w};
#pragma unroll
    for (int i = 0; i < 4; i++) xn[i] = xin[i] * rstd * w[lane * 4 + i];

    // NeoX RoPE: pair (d, d+64); partner lives in lane^16 at same i.
    const float pf = (float)pos[t];
    Pack4 p;
#pragma unroll
    for (int i = 0; i < 4; i++) {
        const int d  = lane * 4 + i;
        const float other = __shfl_xor(xn[i], 16, 32);
        const int fi = d & 63;
        const float th = pf * __powf(10000.0f, -(float)fi * (1.0f / 64.0f));
        const float c = __cosf(th), s = __sinf(th);
        const float res = (d < 64) ? (xn[i] * c - other * s)
                                   : (xn[i] * c + other * s);
        p.h[i] = (__bf16)res;
    }
    __bf16* dst = (slot < 32)
        ? qb + ((size_t)slot * TT + t) * HD + lane * 4
        : kb + ((size_t)(slot - 32) * TT + t) * HD + lane * 4;
    *reinterpret_cast<uint2*>(dst) = p.u2;
}

// =====================================================================
// Flash attention: 1 wave per (head, 16-row q block). Causal, GQA 4:1.
// S = Q*K^T via 16x16x32 WMMA; online softmax in registers; P bounced
// through LDS into A-fragment layout; V chunk async-DMA'd into LDS.
// Output bf16 [T][NH*HD].
// =====================================================================
__global__ __launch_bounds__(32) void flash_attn(
    const __bf16* __restrict__ Q, const __bf16* __restrict__ Kh,
    const __bf16* __restrict__ V, __bf16* __restrict__ O)
{
    __shared__ __bf16 sP[16][32];    // probabilities tile (row-major)
    __shared__ __bf16 sV[32][128];   // V chunk, row-major [k][d]

    const int lane = threadIdx.x;
    const int r    = lane & 15;
    const int kg   = lane >> 4;
    const int h    = blockIdx.x >> 7;     // / (T/16)
    const int qb   = blockIdx.x & 127;
    const int kvh  = h >> 2;              // GQA group of 4
    const int q0   = qb * 16;
    const float scale = 0.08838834764831845f;  // 128^-0.5

    // ---- Q fragments (16x128 -> 4 A-frags), direct from global ----
    Frag16 aQ[4];
    const unsigned* gQ = reinterpret_cast<const unsigned*>(Q + ((size_t)h * TT + q0) * HD);
#pragma unroll
    for (int kk = 0; kk < 4; kk++)
#pragma unroll
        for (int j = 0; j < 4; j++) {
            aQ[kk].u[j]     = gQ[r * 64 + kk * 16 + kg * 4 + j];
            aQ[kk].u[4 + j] = gQ[r * 64 + kk * 16 + 8 + kg * 4 + j];
        }

    float m_i[8], l_i[8];
    v8f o[8];
#pragma unroll
    for (int i = 0; i < 8; i++) {
        m_i[i] = -3.0e38f; l_i[i] = 0.f;
        o[i] = v8f{0.f,0.f,0.f,0.f,0.f,0.f,0.f,0.f};
    }

    const unsigned* gK = reinterpret_cast<const unsigned*>(Kh + (size_t)kvh * TT * HD);
    const __bf16*   gV = V + (size_t)kvh * TT * HD;

    for (int kc = 0; kc < q0 + 16; kc += 32) {
        // ---- async-DMA V chunk (32x128 bf16 = 8KB) into LDS row-major ----
        asm volatile("s_wait_dscnt 0x0" ::: "memory");  // prior chunk's LDS reads done
#pragma unroll
        for (int i = 0; i < 16; i++) {
            const int off = (i * 32 + lane) * 8;        // 0..4095 elements
            async_ld_b128(gV + (size_t)kc * HD + off, reinterpret_cast<__bf16*>(sV) + off);
        }

        // ---- S tiles: keys [kc..kc+15] and [kc+16..kc+31] ----
        v8f s0 = v8f{0.f,0.f,0.f,0.f,0.f,0.f,0.f,0.f};
        v8f s1 = s0;
#pragma unroll
        for (int kk = 0; kk < 4; kk++) {
            Frag16 b0, b1;
            const unsigned* rowK0 = gK + (size_t)(kc + r) * 64;
            const unsigned* rowK1 = gK + (size_t)(kc + 16 + r) * 64;
#pragma unroll
            for (int i = 0; i < 8; i++) {
                b0.u[i] = rowK0[kk * 16 + (i >> 2) * 8 + kg * 4 + (i & 3)];
                b1.u[i] = rowK1[kk * 16 + (i >> 2) * 8 + kg * 4 + (i & 3)];
            }
            s0 = wmma_bf16(aQ[kk], b0, s0);
            s1 = wmma_bf16(aQ[kk], b1, s1);
        }

        // ---- online softmax (rows rr+8*kg live in one 16-lane half) ----
#pragma unroll
        for (int rr = 0; rr < 8; rr++) {
            const int row = q0 + rr + 8 * kg;
            const int c0 = kc + r, c1 = kc + 16 + r;
            float v0 = s0[rr] * scale, v1 = s1[rr] * scale;
            if (c0 > row) v0 = -3.0e38f;
            if (c1 > row) v1 = -3.0e38f;
            float mx = fmaxf(v0, v1);
#pragma unroll
            for (int m = 1; m < 16; m <<= 1) mx = fmaxf(mx, __shfl_xor(mx, m, 32));
            const float mnew = fmaxf(m_i[rr], mx);
            const float corr = __expf(m_i[rr] - mnew);
            const float p0 = (c0 > row) ? 0.f : __expf(v0 - mnew);
            const float p1 = (c1 > row) ? 0.f : __expf(v1 - mnew);
            float ps = p0 + p1;
#pragma unroll
            for (int m = 1; m < 16; m <<= 1) ps += __shfl_xor(ps, m, 32);
            l_i[rr] = l_i[rr] * corr + ps;
            m_i[rr] = mnew;
#pragma unroll
            for (int j = 0; j < 8; j++) o[j][rr] *= corr;
            sP[rr + 8 * kg][r]      = (__bf16)p0;
            sP[rr + 8 * kg][16 + r] = (__bf16)p1;
        }
        asm volatile("s_wait_dscnt 0x0" ::: "memory");

        // ---- P as A-fragment from LDS ----
        Frag16 aP;
        const unsigned* lP = reinterpret_cast<const unsigned*>(sP);  // stride 16 u32
#pragma unroll
        for (int j = 0; j < 4; j++) {
            aP.u[j]     = lP[r * 16 + kg * 4 + j];
            aP.u[4 + j] = lP[r * 16 + 8 + kg * 4 + j];
        }

        // ---- O += P(16x32) * Vchunk(32x128) ----
        wait_async0();   // V chunk landed in LDS
#pragma unroll
        for (int j = 0; j < 8; j++) {
            Frag16 bv;
            const int d = j * 16 + r;
#pragma unroll
            for (int i = 0; i < 8; i++) {
                const int k0 = (i >> 2) * 16 + kg * 8 + 2 * (i & 3);          // k-pair spans two sV rows
                Pack2 pk;
                pk.h[0] = sV[k0][d];
                pk.h[1] = sV[k0 + 1][d];
                bv.u[i] = pk.u;
            }
            o[j] = wmma_bf16(aP, bv, o[j]);
        }
    }

    // ---- epilogue: divide by l, store bf16 [t][h*128+d] ----
#pragma unroll
    for (int rr = 0; rr < 8; rr++) {
        const float rl = 1.0f / l_i[rr];
        const int trow = q0 + rr + 8 * kg;
        __bf16* dst = O + (size_t)trow * (NH * HD) + h * HD + r;
#pragma unroll
        for (int j = 0; j < 8; j++)
            dst[j * 16] = (__bf16)(o[j][rr] * rl);
    }
}

// =====================================================================
// launch
// =====================================================================
extern "C" void kernel_launch(void* const* d_in, const int* in_sizes, int n_in,
                              void* d_out, int out_size, void* d_ws, size_t ws_size,
                              hipStream_t stream) {
    const float*     hs   = (const float*)d_in[0];
    const long long* pos  = (const long long*)d_in[1];   // int64 positions
    const float*     wqkv = (const float*)d_in[2];
    const float*     wo   = (const float*)d_in[3];
    const float*     qw   = (const float*)d_in[4];
    const float*     kw   = (const float*)d_in[5];
    float*           out  = (float*)d_out;

    // workspace layout
    __bf16* hsb   = (__bf16*)d_ws;                                  // 2048x4096
    __bf16* wqkvb = hsb   + (size_t)TT * HIDDEN;                    // 6144x4096
    __bf16* wob   = wqkvb + (size_t)QKV_OUT * HIDDEN;               // 4096x4096
    float*  qkv   = (float*)(wob + (size_t)HIDDEN * HIDDEN);        // 2048x6144 f32
    __bf16* qb    = (__bf16*)(qkv + (size_t)TT * QKV_OUT);          // [32][2048][128]
    __bf16* kb    = qb + (size_t)NH  * TT * HD;                     // [8][2048][128]
    __bf16* vb    = kb + (size_t)NKV * TT * HD;                     // [8][2048][128]
    __bf16* attnb = vb + (size_t)NKV * TT * HD;                     // 2048x4096

    // 1) fp32 -> bf16
    cvt_f32_bf16<<<4096, 256, 0, stream>>>(hs,   hsb,   TT * HIDDEN / 4);
    cvt_f32_bf16<<<4096, 256, 0, stream>>>(wqkv, wqkvb, QKV_OUT * HIDDEN / 4);
    cvt_f32_bf16<<<4096, 256, 0, stream>>>(wo,   wob,   HIDDEN * HIDDEN / 4);

    // 2) QKV projection: (2048x4096)x(6144x4096)^T
    gemm_bf16_nt<<<dim3(QKV_OUT / BN, TT / BM), 256, 0, stream>>>(
        hsb, wqkvb, qkv, TT, QKV_OUT, HIDDEN);

    // 3) fused RMSNorm + RoPE + pack
    norm_rope_bf16<<<(TT * 48) / 4, 128, 0, stream>>>(qkv, pos, qw, kw, qb, kb, vb);

    // 4) flash attention: 32 heads x 128 q-blocks, one wave each
    flash_attn<<<NH * (TT / 16), 32, 0, stream>>>(qb, kb, vb, attnb);

    // 5) output projection: (2048x4096)x(4096x4096)^T -> f32 out
    gemm_bf16_nt<<<dim3(HIDDEN / BN, TT / BM), 256, 0, stream>>>(
        attnb, wob, out, TT, HIDDEN, HIDDEN);
}
